// LanguageModel_45079976738861
// MI455X (gfx1250) — compile-verified, matrix operand order checked by reading, not run on hardware
//
#include <hip/hip_runtime.h>
#include <hip/hip_bf16.h>

// Problem constants (match reference)
#define BB 512
#define TT 8
#define VV 50257
#define DD 1024
#define HH 4
#define HD 256

typedef __attribute__((ext_vector_type(16))) __bf16 v16bf;
typedef __attribute__((ext_vector_type(8)))  float  v8f;

struct FragBits { uint4 lo, hi; };

// ---------------------------------------------------------------------------
// CDNA5 async global->LDS copy (ASYNCcnt-tracked, no VGPR staging).
// Generic pointers to __shared__ truncate to the LDS byte offset (ISA 10.2).
// ---------------------------------------------------------------------------
__device__ __forceinline__ void async_ld_b128(__hip_bfloat16* lds,
                                              const __hip_bfloat16* g) {
  unsigned l32 = (unsigned)(unsigned long long)lds;
  asm volatile("global_load_async_to_lds_b128 %0, %1, off"
               :: "v"(l32), "v"(g) : "memory");
}
__device__ __forceinline__ void wait_async_4() {
  asm volatile("s_wait_asynccnt 4" ::: "memory");
}
__device__ __forceinline__ void wait_async_0() {
  asm volatile("s_wait_asynccnt 0" ::: "memory");
}

// Load one 16x32 bf16 WMMA operand fragment from an LDS tile with row stride
// LSTR bf16 elements. Layout per ISA 7.12.2 (A and dense-B patterns match):
// lane l -> row/col (l&15); v0..v3 = K {base..base+7}, v4..v7 = K {16+base..},
// base = (l>=16)*8.
template<int LSTR>
__device__ __forceinline__ v16bf lds_frag(const __hip_bfloat16* tile,
                                          int row, int koff) {
  const __hip_bfloat16* p = tile + row * LSTR + koff;
  FragBits f;
  f.lo = *reinterpret_cast<const uint4*>(p);        // k .. k+7
  f.hi = *reinterpret_cast<const uint4*>(p + 16);   // k+16 .. k+23
  return __builtin_bit_cast(v16bf, f);
}

// ---------------------------------------------------------------------------
// C[M][N] = A[M][K] * B[N][K]^T (+ bias[N]); A,B bf16, C f32.
// Block = 8 waves (4Mx2N). Block tile 128x128, wave tile 32x64 (2x4 WMMA
// sub-tiles). K-steps of 32 double-buffered in LDS via async copies.
// ---------------------------------------------------------------------------
template<int KK>   // K (compile-time; 1024 here)
__global__ __launch_bounds__(256)
void wmma_gemm_lds(const __hip_bfloat16* __restrict__ A,
                   const __hip_bfloat16* __restrict__ Bm,
                   const float* __restrict__ bias,
                   float* __restrict__ C,
                   int M, int N, int blocksN)
{
  constexpr int TM = 2, TN = 4;          // wave sub-tiles
  constexpr int BM = 128, BN = 128;      // block tile
  constexpr int LSTR = 40;               // LDS row stride (bf16): 80B, pad kills bank conflicts
  constexpr int KSTEPS = KK / 32;

  __shared__ __hip_bfloat16 sA[2][BM][LSTR];
  __shared__ __hip_bfloat16 sB[2][BN][LSTR];

  const int tid  = threadIdx.x;
  const int lane = tid & 31;
  const int wave = tid >> 5;
  const int wm   = wave >> 1;            // 0..3
  const int wn   = wave & 1;             // 0..1

  const int bm0 = (blockIdx.x / blocksN) * BM;
  const int bn0 = (blockIdx.x % blocksN) * BN;

  const int r    = lane & 15;            // row (A) / col (B) within 16x16 tile
  const int koff = (lane >> 4) << 3;     // 0 or 8 bf16 elements
  const int tmW  = wm * 16 * TM;         // wave origin within block tile
  const int tnW  = wn * 16 * TN;

  v8f acc[TM][TN] = {};

  // Stage one 32-wide K slice of A (BM x 32) and B (BN x 32) into buffer buf.
  // 512 + 512 b128 chunks, 4 per thread -> ASYNCcnt 4 per stage per thread.
  auto stage = [&](int buf, int kk) {
#pragma unroll
    for (int c = tid; c < BM * 4; c += 256) {       // 4 x 16B chunks per row
      const int row = c >> 2, seg = c & 3;
      async_ld_b128(&sA[buf][row][seg * 8],
                    A + (size_t)(bm0 + row) * KK + kk + seg * 8);
    }
#pragma unroll
    for (int c = tid; c < BN * 4; c += 256) {
      const int row = c >> 2, seg = c & 3;
      int rn = bn0 + row;
      if (rn >= N) rn = N - 1;                      // clamp; stores guarded
      async_ld_b128(&sB[buf][row][seg * 8],
                    Bm + (size_t)rn * KK + kk + seg * 8);
    }
  };

  auto compute = [&](int buf) {
    v16bf af[TM], bfv[TN];
#pragma unroll
    for (int i = 0; i < TM; ++i)
      af[i] = lds_frag<LSTR>(&sA[buf][0][0], tmW + i * 16 + r, koff);
#pragma unroll
    for (int j = 0; j < TN; ++j)
      bfv[j] = lds_frag<LSTR>(&sB[buf][0][0], tnW + j * 16 + r, koff);
#pragma unroll
    for (int i = 0; i < TM; ++i)
#pragma unroll
      for (int j = 0; j < TN; ++j)
        acc[i][j] = __builtin_amdgcn_wmma_f32_16x16x32_bf16(
            false, af[i], false, bfv[j], (short)0, acc[i][j], false, false);
  };

  stage(0, 0);
  for (int ks = 0; ks < KSTEPS - 1; ++ks) {
    stage((ks & 1) ^ 1, (ks + 1) * 32);   // prefetch next slice
    wait_async_4();                       // current slice resident (4 newer pending)
    __syncthreads();
    compute(ks & 1);
    __syncthreads();                      // LDS reads done before next overwrite
  }
  wait_async_0();
  __syncthreads();
  compute((KSTEPS - 1) & 1);

  // Epilogue. C/D layout: VGPR e, lanes 0-15 -> M=e, lanes 16-31 -> M=8+e.
  const int rowOff = (lane >> 4) * 8;
#pragma unroll
  for (int i = 0; i < TM; ++i) {
#pragma unroll
    for (int j = 0; j < TN; ++j) {
      const int col = bn0 + tnW + j * 16 + (lane & 15);
      if (col < N) {
        const float bv = bias ? bias[col] : 0.0f;
#pragma unroll
        for (int e = 0; e < 8; ++e) {
          const int row = bm0 + tmW + i * 16 + rowOff + e;
          C[(size_t)row * N + col] = acc[i][j][e] + bv;
        }
      }
    }
  }
}

// ---------------------------------------------------------------------------
// h[b,t,:] = tok_emb[x[b,t]] + pos_emb[t], emitted as bf16.
// ---------------------------------------------------------------------------
__global__ void embed_kernel(const int* __restrict__ x,
                             const float* __restrict__ tok,
                             const float* __restrict__ pos,
                             __hip_bfloat16* __restrict__ hB)
{
  const int bt  = blockIdx.x;           // 0 .. B*T-1
  const int t   = bt % TT;
  const int tid = x[bt];
  const float* tr = tok + (size_t)tid * DD;
  const float* pr = pos + (size_t)t * DD;
  for (int i = threadIdx.x; i < DD; i += blockDim.x)
    hB[(size_t)bt * DD + i] = __float2bfloat16(tr[i] + pr[i]);
}

// ---------------------------------------------------------------------------
// Generic f32 -> bf16 conversion (grid-stride).
// ---------------------------------------------------------------------------
__global__ void cvt_bf16_kernel(const float* __restrict__ in,
                                __hip_bfloat16* __restrict__ out, size_t n)
{
  for (size_t i = (size_t)blockIdx.x * blockDim.x + threadIdx.x; i < n;
       i += (size_t)gridDim.x * blockDim.x)
    out[i] = __float2bfloat16(in[i]);
}

// ---------------------------------------------------------------------------
// Attention: one 64-thread block per (b,h). q/k/v are f32 in "[bt][h*HD+d]"
// layout (direct GEMM output). Causal softmax, scale = 1/sqrt(256) = 1/16.
// Writes head-concatenated output directly as bf16 [bt][D].
// ---------------------------------------------------------------------------
__global__ __launch_bounds__(64)
void attn_kernel(const float* __restrict__ q,
                 const float* __restrict__ k,
                 const float* __restrict__ v,
                 __hip_bfloat16* __restrict__ out)
{
  __shared__ float sq[TT][HD], sk[TT][HD], sv[TT][HD];
  __shared__ float w[TT][TT];

  const int bh = blockIdx.x;
  const int b  = bh / HH;
  const int h  = bh % HH;
  const int tid = threadIdx.x;

  for (int i = tid; i < TT * HD; i += 64) {
    const int t = i / HD, d = i % HD;
    const size_t idx = (size_t)(b * TT + t) * DD + h * HD + d;
    sq[t][d] = q[idx];
    sk[t][d] = k[idx];
    sv[t][d] = v[idx];
  }
  __syncthreads();

  {   // 64 threads -> 64 (t,s) score pairs
    const int t = tid >> 3, s = tid & 7;
    float acc = 0.0f;
    for (int d = 0; d < HD; ++d) acc += sq[t][d] * sk[s][d];
    w[t][s] = (s <= t) ? acc * 0.0625f : -1e9f;
  }
  __syncthreads();

  if (tid < TT) {   // per-row softmax
    float mx = -1e30f;
    for (int s = 0; s < TT; ++s) mx = fmaxf(mx, w[tid][s]);
    float sum = 0.0f;
    for (int s = 0; s < TT; ++s) { float e = __expf(w[tid][s] - mx); w[tid][s] = e; sum += e; }
    const float inv = 1.0f / sum;
    for (int s = 0; s < TT; ++s) w[tid][s] *= inv;
  }
  __syncthreads();

  for (int i = tid; i < TT * HD; i += 64) {
    const int t = i / HD, d = i % HD;
    float acc = 0.0f;
    for (int s = 0; s < TT; ++s) acc += w[t][s] * sv[s][d];
    out[(size_t)(b * TT + t) * DD + h * HD + d] = __float2bfloat16(acc);
  }
}

// ---------------------------------------------------------------------------
extern "C" void kernel_launch(void* const* d_in, const int* in_sizes, int n_in,
                              void* d_out, int out_size, void* d_ws, size_t ws_size,
                              hipStream_t stream)
{
  (void)in_sizes; (void)n_in; (void)out_size; (void)ws_size;

  const int*   x    = (const int*)  d_in[0];
  const float* tok  = (const float*)d_in[1];
  const float* pos  = (const float*)d_in[2];
  const float* Wq   = (const float*)d_in[3];
  const float* Wk   = (const float*)d_in[4];
  const float* Wv   = (const float*)d_in[5];
  const float* Wll  = (const float*)d_in[6];
  const float* bll  = (const float*)d_in[7];
  float* logits     = (float*)d_out;

  // Workspace carve-up (256B aligned)
  size_t off = 0;
  auto carve = [&](size_t bytes) -> char* {
    char* p = (char*)d_ws + off;
    off += (bytes + 255) & ~(size_t)255;
    return p;
  };
  const size_t MT = (size_t)BB * TT;                         // 4096 tokens
  __hip_bfloat16* hB    = (__hip_bfloat16*)carve(MT * DD * 2);
  __hip_bfloat16* WqB   = (__hip_bfloat16*)carve((size_t)DD * DD * 2);
  __hip_bfloat16* WkB   = (__hip_bfloat16*)carve((size_t)DD * DD * 2);
  __hip_bfloat16* WvB   = (__hip_bfloat16*)carve((size_t)DD * DD * 2);
  __hip_bfloat16* WllB  = (__hip_bfloat16*)carve((size_t)VV * DD * 2);  // 103 MB: L2-resident
  float*          qF    = (float*)carve(MT * DD * 4);
  float*          kF    = (float*)carve(MT * DD * 4);
  float*          vF    = (float*)carve(MT * DD * 4);
  __hip_bfloat16* attnB = (__hip_bfloat16*)carve(MT * DD * 2);

  // 1) embeddings -> bf16
  embed_kernel<<<(int)MT, 256, 0, stream>>>(x, tok, pos, hB);

  // 2) weight conversions f32 -> bf16
  cvt_bf16_kernel<<<1024, 256, 0, stream>>>(Wq,  WqB,  (size_t)DD * DD);
  cvt_bf16_kernel<<<1024, 256, 0, stream>>>(Wk,  WkB,  (size_t)DD * DD);
  cvt_bf16_kernel<<<1024, 256, 0, stream>>>(Wv,  WvB,  (size_t)DD * DD);
  cvt_bf16_kernel<<<8192, 256, 0, stream>>>(Wll, WllB, (size_t)VV * DD);

  // 3) Q/K/V projections: [4096,1024] x [1024,1024]^T  (block tile 128x128)
  {
    const int M = (int)MT, N = DD;
    const int blocksM = M / 128, blocksN = (N + 127) / 128;
    const int grid = blocksM * blocksN;
    wmma_gemm_lds<DD><<<grid, 256, 0, stream>>>(hB, WqB, nullptr, qF, M, N, blocksN);
    wmma_gemm_lds<DD><<<grid, 256, 0, stream>>>(hB, WkB, nullptr, kF, M, N, blocksN);
    wmma_gemm_lds<DD><<<grid, 256, 0, stream>>>(hB, WvB, nullptr, vF, M, N, blocksN);
  }

  // 4) causal attention per (b,h)
  attn_kernel<<<BB * HH, 64, 0, stream>>>(qF, kF, vF, attnB);

  // 5) logits: [4096,1024] x [1024,50257]^T + bll  (the 422 GFLOP GEMM)
  {
    const int M = (int)MT, N = VV;
    const int blocksM = M / 128, blocksN = (N + 127) / 128;   // 32 x 393
    const int grid = blocksM * blocksN;
    wmma_gemm_lds<DD><<<grid, 256, 0, stream>>>(attnB, WllB, bll, logits, M, N, blocksN);
  }
}
